// BahdanauEncoder_18940805775589
// MI455X (gfx1250) — matrix operand order repaired and checked by reading, not compiled, not run on hardware
//
#include <hip/hip_runtime.h>
#include <hip/hip_bf16.h>
#include <math.h>

// Problem constants (from reference): V=50000, D=300, H=256, B=32, LA=LC=256
#define BN   32
#define HN   256
#define DN   300
#define DPAD 320      // D padded to multiple of 32 for WMMA K-chunks
#define G3   768      // 3*H
#define LA_N 256
#define LC_N 256

typedef __attribute__((ext_vector_type(16))) __bf16 bf16x16;
typedef __attribute__((ext_vector_type(8)))  __bf16 bf16x8;
typedef __attribute__((ext_vector_type(8)))  float  floatx8;

#ifndef __has_builtin
#define __has_builtin(x) 0
#endif
#if __has_builtin(__builtin_amdgcn_tensor_load_to_lds) && \
    __has_builtin(__builtin_amdgcn_s_wait_tensorcnt)
#define HAVE_TDM 1
#endif

// ---------------------------------------------------------------------------
// WMMA helpers (CDNA5 v_wmma_f32_16x16x32_bf16, wave32)
// ---------------------------------------------------------------------------
__device__ __forceinline__ floatx8 wmma_bf16(bf16x16 a, bf16x16 b, floatx8 c) {
  // 8 args: (neg_a, A, neg_b, B, c_mod, C, reuse_a, reuse_b)
  return __builtin_amdgcn_wmma_f32_16x16x32_bf16(false, a, false, b, (short)0, c,
                                                 false, false);
}

// A-matrix 16x32 bf16, row-major source with leading dim `lda`.
// ISA 7.12.2: lanes 0-15 -> M=lane, K halves {0..7,16..23}; lanes 16-31 -> {8..15,24..31}
__device__ __forceinline__ bf16x16 load_a16x32(const __bf16* base, int lda, int lane) {
  const int row = lane & 15;
  const int kb  = (lane >> 4) << 3;   // 0 or 8
  const __bf16* p = base + row * lda;
  bf16x8 lo = *(const bf16x8*)(p + kb);
  bf16x8 hi = *(const bf16x8*)(p + kb + 16);
  bf16x16 r;
#pragma unroll
  for (int i = 0; i < 8; ++i) { r[i] = lo[i]; r[i + 8] = hi[i]; }
  return r;
}

// B-matrix 32x16 bf16 supplied as B^T row-major ([N][K], k contiguous, ld=ldb).
__device__ __forceinline__ bf16x16 load_bt16x32(const __bf16* base, int ldb, int lane) {
  const int n  = lane & 15;
  const int kb = (lane >> 4) << 3;    // 0 or 8
  const __bf16* p = base + n * ldb;
  bf16x8 lo = *(const bf16x8*)(p + kb);
  bf16x8 hi = *(const bf16x8*)(p + kb + 16);
  bf16x16 r;
#pragma unroll
  for (int i = 0; i < 8; ++i) { r[i] = lo[i]; r[i + 8] = hi[i]; }
  return r;
}

#ifdef HAVE_TDM
typedef __attribute__((ext_vector_type(4))) unsigned int uint32x4;
typedef __attribute__((ext_vector_type(8))) int          int32x8;
typedef __attribute__((ext_vector_type(4))) int          int32x4;

// 1-row TDM descriptor: contiguous nElems fp32 from global -> LDS (ISA ch.8)
__device__ __forceinline__ void tdm_load_1d(unsigned lds_off, const void* gptr,
                                            unsigned nElems) {
  unsigned long long ga = (unsigned long long)gptr;
  uint32x4 g0;
  g0[0] = 1u;                                              // count=1, user desc
  g0[1] = lds_off;                                         // lds_addr (bytes)
  g0[2] = (unsigned)(ga & 0xffffffffu);                    // global_addr[31:0]
  g0[3] = (unsigned)((ga >> 32) & 0x01ffffffu) | (2u << 30); // ga[56:32] | type=2
  int32x8 g1;
  g1[0] = (int)(2u << 16);                                 // data_size=4B; mask=0
  g1[1] = (int)((nElems & 0xffffu) << 16);                 // tensor_dim0[15:0]
  g1[2] = (int)((nElems >> 16) & 0xffffu) | (int)(1u << 16); // dim0[31:16], dim1=1
  g1[3] = (int)((nElems & 0xffffu) << 16);                 // tile_dim0 (<=65535)
  g1[4] = 1;                                               // tile_dim1=1
  g1[5] = (int)nElems;                                     // dim0_stride[31:0]
  g1[6] = 0; g1[7] = 0;
  int32x4 z4 = {0, 0, 0, 0};
#if __clang_major__ >= 23
  int32x8 z8 = {0, 0, 0, 0, 0, 0, 0, 0};
  __builtin_amdgcn_tensor_load_to_lds(g0, g1, z4, z4, z8, 0);
#else
  __builtin_amdgcn_tensor_load_to_lds(g0, g1, z4, z4, 0);
#endif
}
#endif

// ---------------------------------------------------------------------------
// Weight preparation
// ---------------------------------------------------------------------------
__global__ void k_convert_pad(const float* __restrict__ src, __bf16* __restrict__ dst,
                              int N, int Ksrc, int Kdst, int srcLd, int srcOff) {
  int i = blockIdx.x * blockDim.x + threadIdx.x;
  if (i >= N * Kdst) return;
  int n = i / Kdst, k = i % Kdst;
  dst[i] = (k < Ksrc) ? (__bf16)src[(size_t)n * srcLd + srcOff + k] : (__bf16)0.0f;
}

__global__ void k_convert_t(const float* __restrict__ src, __bf16* __restrict__ dst,
                            int N, int K) {
  int i = blockIdx.x * blockDim.x + threadIdx.x;
  if (i >= N * K) return;
  int n = i / K, k = i % K;
  dst[i] = (__bf16)src[(size_t)k * N + n];   // BT[n][k] = B[k][n]
}

// Embedding gather: dst[(t*B+b)*DPAD + k] (zero-padded K)
__global__ void k_embed(const float* __restrict__ emb, const int* __restrict__ ids,
                        int L, __bf16* __restrict__ dst) {
  long i = (long)blockIdx.x * blockDim.x + threadIdx.x;
  if (i >= (long)L * BN * DPAD) return;
  int k  = (int)(i % DPAD);
  long tb = i / DPAD;
  int b  = (int)(tb % BN);
  int t  = (int)(tb / BN);
  float v = 0.0f;
  if (k < DN) v = emb[(long)ids[b * L + t] * DN + k];
  dst[i] = (__bf16)v;
}

// ---------------------------------------------------------------------------
// Generic WMMA GEMM, register-blocked 1 M-tile x 4 N-tiles per wave:
// A fragment loaded once, reused across 4 WMMAs.
// ---------------------------------------------------------------------------
__global__ void k_gemm_bt(const __bf16* __restrict__ A, int lda,
                          const __bf16* __restrict__ BT, int ldb,
                          const float* __restrict__ bias,
                          float* __restrict__ C, int ldc,
                          int Mtiles, int Ngrp, int Kch) {   // Ngrp = Ntiles/4
  const int lane = threadIdx.x & 31;
  const int wave = (int)((blockIdx.x * blockDim.x + threadIdx.x) >> 5);
  if (wave >= Mtiles * Ngrp) return;            // wave-uniform: EXEC stays all-ones
  const int mt = wave / Ngrp, nt0 = (wave % Ngrp) * 4;
  const __bf16* Ap = A + (size_t)mt * 16 * lda;
  floatx8 c[4] = {};
  for (int kc = 0; kc < Kch; ++kc) {
    bf16x16 a = load_a16x32(Ap + kc * 32, lda, lane);
#pragma unroll
    for (int u = 0; u < 4; ++u)
      c[u] = wmma_bf16(a, load_bt16x32(BT + (size_t)(nt0 + u) * 16 * ldb + kc * 32,
                                       ldb, lane), c[u]);
  }
  const int rbase = (lane >> 4) << 3;           // C/D: row = vgpr + 8*(lane>=16)
  const int col   = lane & 15;
#pragma unroll
  for (int u = 0; u < 4; ++u) {
    float bv = bias ? bias[(nt0 + u) * 16 + col] : 0.0f;
    float* Cp = C + (size_t)(mt * 16) * ldc + (nt0 + u) * 16 + col;
#pragma unroll
    for (int r = 0; r < 8; ++r) Cp[(size_t)(rbase + r) * ldc] = c[u][r] + bv;
  }
}

// ---------------------------------------------------------------------------
// Anchor GRU scan: persistent single workgroup (32 waves), h lives in LDS.
// TDM stages gi[t] (96 KB) into LDS, overlapped with the h@W_hh WMMA phase.
// ---------------------------------------------------------------------------
__global__ void __launch_bounds__(1024)
k_anchor_scan(const float* __restrict__ gi,       // [LA][B][768] (x@W_ih^T + b_ih)
              const __bf16* __restrict__ whh,     // [768][256] BT
              const float* __restrict__ bhh,      // [768]
              const int* __restrict__ alen,       // [B]
              __bf16* __restrict__ a_out,         // [LA][B][256] bf16
              __bf16* __restrict__ h_fin,         // [B][256] bf16
              float* __restrict__ gh_scr,         // [B][768] L2-backed scratch
              float* __restrict__ anchor_emb) {   // [B][256] -> d_out
  __shared__ __align__(32) __bf16 h_bf[BN * HN];  // 16 KB recurrent state
  __shared__ __align__(32) float  ssum[BN * HN];  // 32 KB running mean numerator
#ifdef HAVE_TDM
  __shared__ __align__(32) float  gi_lds[BN * G3];// 96 KB staged gate slab
#endif
  const int tid = threadIdx.x;
  const int lane = tid & 31;
  const int wave = tid >> 5;
  for (int i = tid; i < BN * HN; i += 1024) { h_bf[i] = (__bf16)0.0f; ssum[i] = 0.0f; }
  __syncthreads();

  for (int t = 0; t < LA_N; ++t) {
#ifdef HAVE_TDM
    if (wave == 0)   // async DMA of this step's precomputed gates; overlaps GEMM
      tdm_load_1d((unsigned)(unsigned long long)(void*)gi_lds,
                  gi + (size_t)t * BN * G3, BN * G3);
#endif
    // gh = h @ W_hh^T : 2x48 tiles; wave -> (mt, 3 consecutive nt), A reused x3
    {
      int mt  = wave & 1;
      int nt0 = (wave >> 1) * 3;
      const __bf16* Ap = h_bf + mt * 16 * HN;
      floatx8 c[3] = {};
      for (int kc = 0; kc < 8; ++kc) {
        bf16x16 a = load_a16x32(Ap + kc * 32, HN, lane);
#pragma unroll
        for (int u = 0; u < 3; ++u)
          c[u] = wmma_bf16(a, load_bt16x32(whh + (size_t)(nt0 + u) * 16 * HN + kc * 32,
                                           HN, lane), c[u]);
      }
      int rbase = (lane >> 4) << 3, col = lane & 15;
#pragma unroll
      for (int u = 0; u < 3; ++u) {
        float* Cp = gh_scr + (size_t)(mt * 16) * G3 + (nt0 + u) * 16 + col;
#pragma unroll
        for (int r = 0; r < 8; ++r) Cp[(size_t)(rbase + r) * G3] = c[u][r];
      }
    }
    __threadfence();
#ifdef HAVE_TDM
    if (wave == 0) __builtin_amdgcn_s_wait_tensorcnt(0);
#endif
    __syncthreads();

#ifdef HAVE_TDM
    const float* git = gi_lds;
#else
    const float* git = gi + (size_t)t * BN * G3;
    if (t + 1 < LA_N)
      __builtin_prefetch(git + (size_t)BN * G3 + tid * 16, 0, 1);
#endif
#pragma unroll
    for (int i = 0; i < 8; ++i) {
      int id = tid + i * 1024;
      int b = id >> 8, j = id & 255;
      float ir = git[b * G3 + j];
      float iz = git[b * G3 + 256 + j];
      float in = git[b * G3 + 512 + j];
      float hr = gh_scr[b * G3 + j]       + bhh[j];
      float hz = gh_scr[b * G3 + 256 + j] + bhh[256 + j];
      float hn = gh_scr[b * G3 + 512 + j] + bhh[512 + j];
      float r  = 1.0f / (1.0f + __expf(-(ir + hr)));
      float z  = 1.0f / (1.0f + __expf(-(iz + hz)));
      float nn = tanhf(in + r * hn);
      float hold = (float)h_bf[b * HN + j];
      float hnew = (1.0f - z) * nn + z * hold;
      bool valid = t < alen[b];
      float hsel = valid ? hnew : hold;     // packed-seq: freeze hidden
      float outv = valid ? hnew : 0.0f;     // zero-pad outputs
      h_bf[b * HN + j] = (__bf16)hsel;
      a_out[((size_t)t * BN + b) * HN + j] = (__bf16)outv;
      ssum[b * HN + j] += outv;
    }
    __syncthreads();
  }
#pragma unroll
  for (int i = 0; i < 8; ++i) {
    int id = tid + i * 1024;
    int b = id >> 8, j = id & 255;
    anchor_emb[b * HN + j] = ssum[b * HN + j] / (float)alen[b];
    h_fin[b * HN + j] = h_bf[b * HN + j];
  }
}

// ---------------------------------------------------------------------------
// Candidate scan: attention + GRU per step, persistent single workgroup
// ---------------------------------------------------------------------------
__global__ void __launch_bounds__(1024)
k_cand_scan(const float* __restrict__ gix,        // [LC][B][768] x-part (incl b_ih_c)
            const __bf16* __restrict__ whh,       // [768][256] BT (W_hh_c)
            const __bf16* __restrict__ wcc,       // [768][256] BT (ctx cols of W_ih_c)
            const __bf16* __restrict__ wq,        // [256][256] BT (Wq^T)
            const float* __restrict__ bhh,        // [768]
            const float* __restrict__ vatt,       // [256]
            const int* __restrict__ clen,         // [B]
            const float* __restrict__ keys,       // [LA][B][256] fp32
            const __bf16* __restrict__ a_out,     // [LA][B][256] bf16
            const __bf16* __restrict__ h0,        // [B][256] bf16
            float* __restrict__ gh_scr,           // [B][768] scratch
            float* __restrict__ gic_scr,          // [B][768] scratch
            float* __restrict__ cand_emb) {       // [B][256] -> d_out+8192
  __shared__ __align__(32) __bf16 h_bf[BN * HN];    // 16 KB
  __shared__ __align__(32) __bf16 q_bf[BN * HN];    // 16 KB
  __shared__ __align__(32) __bf16 ctx_bf[BN * HN];  // 16 KB
  __shared__ __align__(32) float  attn[BN * LA_N];  // 32 KB (scores -> softmax in place)
  __shared__ __align__(32) float  ssum[BN * HN];    // 32 KB
  __shared__ __align__(32) float  vs[HN];           // 1 KB
  const int tid = threadIdx.x;
  const int lane = tid & 31;
  const int wave = tid >> 5;
  for (int i = tid; i < BN * HN; i += 1024) { h_bf[i] = h0[i]; ssum[i] = 0.0f; }
  for (int i = tid; i < HN; i += 1024) vs[i] = vatt[i];
  __syncthreads();

  for (int t = 0; t < LC_N; ++t) {
    // q = h @ Wq : 2x16 tiles, one per wave
    {
      int mt = wave >> 4, nt = wave & 15;
      const __bf16* Ap = h_bf + mt * 16 * HN;
      const __bf16* Bp = wq + (size_t)nt * 16 * HN;
      floatx8 c = {};
      for (int kc = 0; kc < 8; ++kc)
        c = wmma_bf16(load_a16x32(Ap + kc * 32, HN, lane),
                      load_bt16x32(Bp + kc * 32, HN, lane), c);
      int rbase = (lane >> 4) << 3, col = lane & 15;
#pragma unroll
      for (int r = 0; r < 8; ++r)
        q_bf[(mt * 16 + rbase + r) * HN + nt * 16 + col] = (__bf16)c[r];
    }
    __syncthreads();

    // scores[b][l] = v . tanh(keys[l][b][:] + q[b][:])
#pragma unroll
    for (int i = 0; i < 8; ++i) {
      int id = tid + i * 1024;
      int b = id >> 8, l = id & 255;
      const float* kp = keys + ((size_t)l * BN + b) * HN;
      const __bf16* qp = q_bf + b * HN;
      float s = 0.0f;
      for (int hh = 0; hh < HN; ++hh)
        s += vs[hh] * tanhf(kp[hh] + (float)qp[hh]);
      attn[b * LA_N + l] = s;
    }
    __syncthreads();

    // softmax over l: wave w handles row b = w (32 lanes x 8 elems)
    {
      int b = wave;
      float loc[8];
      float m = -1e30f;
#pragma unroll
      for (int i = 0; i < 8; ++i) { loc[i] = attn[b * LA_N + lane + i * 32]; m = fmaxf(m, loc[i]); }
#pragma unroll
      for (int off = 16; off > 0; off >>= 1) m = fmaxf(m, __shfl_xor(m, off, 32));
      float sum = 0.0f;
#pragma unroll
      for (int i = 0; i < 8; ++i) { loc[i] = __expf(loc[i] - m); sum += loc[i]; }
#pragma unroll
      for (int off = 16; off > 0; off >>= 1) sum += __shfl_xor(sum, off, 32);
      float inv = 1.0f / sum;
#pragma unroll
      for (int i = 0; i < 8; ++i) attn[b * LA_N + lane + i * 32] = loc[i] * inv;
    }
    __syncthreads();

    // context[b][n] = sum_l attn[b][l] * a_out[l][b][n]
#pragma unroll
    for (int i = 0; i < 8; ++i) {
      int id = tid + i * 1024;
      int b = id >> 8, n = id & 255;
      float acc = 0.0f;
      for (int l = 0; l < LA_N; ++l)
        acc += attn[b * LA_N + l] * (float)a_out[((size_t)l * BN + b) * HN + n];
      ctx_bf[b * HN + n] = (__bf16)acc;
    }
    __syncthreads();

    // gh = h @ W_hh_c^T and gi_ctx = ctx @ W_c^T : 2 matrices x 2 mt x 48 nt
    // wave -> (matrix, mt, 6 consecutive nt); A fragment reused x6
    {
      int sel = wave & 1;
      int mt  = (wave >> 1) & 1;
      int nt0 = (wave >> 2) * 6;
      const __bf16* Ap = (sel ? ctx_bf : h_bf) + mt * 16 * HN;
      const __bf16* W  = sel ? wcc : whh;
      float* Out = sel ? gic_scr : gh_scr;
      floatx8 c[6] = {};
      for (int kc = 0; kc < 8; ++kc) {
        bf16x16 a = load_a16x32(Ap + kc * 32, HN, lane);
#pragma unroll
        for (int u = 0; u < 6; ++u)
          c[u] = wmma_bf16(a, load_bt16x32(W + (size_t)(nt0 + u) * 16 * HN + kc * 32,
                                           HN, lane), c[u]);
      }
      int rbase = (lane >> 4) << 3, col = lane & 15;
#pragma unroll
      for (int u = 0; u < 6; ++u) {
        float* Cp = Out + (size_t)(mt * 16) * G3 + (nt0 + u) * 16 + col;
#pragma unroll
        for (int r = 0; r < 8; ++r) Cp[(size_t)(rbase + r) * G3] = c[u][r];
      }
    }
    __threadfence();
    __syncthreads();

    // gates + length-masked blend
    const float* gxt = gix + (size_t)t * BN * G3;
    if (t + 1 < LC_N)
      __builtin_prefetch(gxt + (size_t)BN * G3 + tid * 16, 0, 1);
#pragma unroll
    for (int i = 0; i < 8; ++i) {
      int id = tid + i * 1024;
      int b = id >> 8, j = id & 255;
      float ir = gxt[b * G3 + j]       + gic_scr[b * G3 + j];
      float iz = gxt[b * G3 + 256 + j] + gic_scr[b * G3 + 256 + j];
      float in = gxt[b * G3 + 512 + j] + gic_scr[b * G3 + 512 + j];
      float hr = gh_scr[b * G3 + j]       + bhh[j];
      float hz = gh_scr[b * G3 + 256 + j] + bhh[256 + j];
      float hn = gh_scr[b * G3 + 512 + j] + bhh[512 + j];
      float r  = 1.0f / (1.0f + __expf(-(ir + hr)));
      float z  = 1.0f / (1.0f + __expf(-(iz + hz)));
      float nn = tanhf(in + r * hn);
      float hold = (float)h_bf[b * HN + j];
      float hnew = (1.0f - z) * nn + z * hold;
      bool frozen = clen[b] < t;            // strict '<', faithful to reference
      float hv = frozen ? hold : hnew;
      h_bf[b * HN + j] = (__bf16)hv;
      if (t < clen[b]) ssum[b * HN + j] += hv;
    }
    __syncthreads();
  }
#pragma unroll
  for (int i = 0; i < 8; ++i) {
    int id = tid + i * 1024;
    int b = id >> 8, j = id & 255;
    cand_emb[b * HN + j] = ssum[b * HN + j] / (float)clen[b];
  }
}

// ---------------------------------------------------------------------------
// Host launcher
// ---------------------------------------------------------------------------
extern "C" void kernel_launch(void* const* d_in, const int* in_sizes, int n_in,
                              void* d_out, int out_size, void* d_ws, size_t ws_size,
                              hipStream_t stream) {
  (void)in_sizes; (void)n_in; (void)out_size; (void)ws_size;
  const int*   a_in   = (const int*)  d_in[0];
  const int*   a_len  = (const int*)  d_in[1];
  const int*   c_in   = (const int*)  d_in[2];
  const int*   c_len  = (const int*)  d_in[3];
  // d_in[4] = batch_size scalar (fixed at 32)
  const float* emb    = (const float*)d_in[5];
  const float* W_ih_a = (const float*)d_in[6];
  const float* W_hh_a = (const float*)d_in[7];
  const float* b_ih_a = (const float*)d_in[8];
  const float* b_hh_a = (const float*)d_in[9];
  const float* W_ih_c = (const float*)d_in[10];
  const float* W_hh_c = (const float*)d_in[11];
  const float* b_ih_c = (const float*)d_in[12];
  const float* b_hh_c = (const float*)d_in[13];
  const float* Wq     = (const float*)d_in[14];
  const float* Wk     = (const float*)d_in[15];
  const float* v_att  = (const float*)d_in[16];
  float* out = (float*)d_out;

  char* ws = (char*)d_ws;
  size_t off = 0;
  auto take = [&](size_t bytes) -> void* {
    void* p = (void*)(ws + off);
    off += (bytes + 255) & ~(size_t)255;
    return p;
  };
  __bf16* xa   = (__bf16*)take((size_t)LA_N * BN * DPAD * 2);   // 5.2 MB
  __bf16* xc   = (__bf16*)take((size_t)LC_N * BN * DPAD * 2);   // 5.2 MB
  __bf16* wiha = (__bf16*)take((size_t)G3 * DPAD * 2);
  __bf16* whha = (__bf16*)take((size_t)G3 * HN * 2);
  __bf16* wicx = (__bf16*)take((size_t)G3 * DPAD * 2);
  __bf16* wicc = (__bf16*)take((size_t)G3 * HN * 2);
  __bf16* whhc = (__bf16*)take((size_t)G3 * HN * 2);
  __bf16* wqT  = (__bf16*)take((size_t)HN * HN * 2);
  __bf16* wkT  = (__bf16*)take((size_t)HN * HN * 2);
  float*  gia  = (float*) take((size_t)LA_N * BN * G3 * 4);     // 25 MB
  float*  gicx = (float*) take((size_t)LC_N * BN * G3 * 4);     // 25 MB
  __bf16* aout = (__bf16*)take((size_t)LA_N * BN * HN * 2);     // 4.2 MB
  float*  keys = (float*) take((size_t)LA_N * BN * HN * 4);     // 8.4 MB
  __bf16* hfin = (__bf16*)take((size_t)BN * HN * 2);
  float*  ghs  = (float*) take((size_t)BN * G3 * 4);
  float*  gics = (float*) take((size_t)BN * G3 * 4);

  auto nb = [](long n, int t) { return (unsigned)((n + t - 1) / t); };

  // --- weight prep (bf16, BT layout; W_ih matrices are already [N][K]) ---
  k_convert_pad<<<nb((long)G3 * DPAD, 256), 256, 0, stream>>>(W_ih_a, wiha, G3, DN, DPAD, DN, 0);
  k_convert_pad<<<nb((long)G3 * HN, 256), 256, 0, stream>>>(W_hh_a, whha, G3, HN, HN, HN, 0);
  k_convert_pad<<<nb((long)G3 * DPAD, 256), 256, 0, stream>>>(W_ih_c, wicx, G3, DN, DPAD, DN + HN, 0);
  k_convert_pad<<<nb((long)G3 * HN, 256), 256, 0, stream>>>(W_ih_c, wicc, G3, HN, HN, DN + HN, DN);
  k_convert_pad<<<nb((long)G3 * HN, 256), 256, 0, stream>>>(W_hh_c, whhc, G3, HN, HN, HN, 0);
  k_convert_t<<<nb((long)HN * HN, 256), 256, 0, stream>>>(Wq, wqT, HN, HN);
  k_convert_t<<<nb((long)HN * HN, 256), 256, 0, stream>>>(Wk, wkT, HN, HN);

  // --- embedding gathers ---
  k_embed<<<nb((long)LA_N * BN * DPAD, 256), 256, 0, stream>>>(emb, a_in, LA_N, xa);
  k_embed<<<nb((long)LC_N * BN * DPAD, 256), 256, 0, stream>>>(emb, c_in, LC_N, xc);

  // --- batched input GEMMs (time dimension is fully parallel) ---
  {
    int mt = LA_N * BN / 16, ng = (G3 / 16) / 4;          // 512 x 12 wave-jobs
    k_gemm_bt<<<nb((long)mt * ng, 8), 256, 0, stream>>>(xa, DPAD, wiha, DPAD, b_ih_a,
                                                        gia, G3, mt, ng, DPAD / 32);
    k_gemm_bt<<<nb((long)mt * ng, 8), 256, 0, stream>>>(xc, DPAD, wicx, DPAD, b_ih_c,
                                                        gicx, G3, mt, ng, DPAD / 32);
  }

  // --- anchor recurrence (writes anchor_emb to d_out[0:8192]) ---
  k_anchor_scan<<<1, 1024, 0, stream>>>(gia, whha, b_hh_a, a_len, aout, hfin, ghs, out);

  // --- keys = anchor_outputs @ Wk ---
  {
    int mt = LA_N * BN / 16, ng = (HN / 16) / 4;          // 512 x 4 wave-jobs
    k_gemm_bt<<<nb((long)mt * ng, 8), 256, 0, stream>>>(aout, HN, wkT, HN, nullptr,
                                                        keys, HN, mt, ng, HN / 32);
  }

  // --- candidate recurrence with attention (writes d_out[8192:16384]) ---
  k_cand_scan<<<1, 1024, 0, stream>>>(gicx, whhc, wicc, wqT, b_hh_c, v_att, c_len,
                                      keys, aout, hfin, ghs, gics, out + BN * HN);
}
